// SparseMoELayer_11948599018368
// MI455X (gfx1250) — matrix operand order repaired and checked by reading, not compiled, not run on hardware
//
#include <hip/hip_runtime.h>
#include <hip/hip_bf16.h>
#include <math.h>

// ---------------- problem constants (match reference init_kwargs) ------------
#define T_TOK   (16 * 8192)   // B*S tokens
#define DMODEL  256
#define DHID    512
#define NEXP    8
#define TOPK    2
#define TILE_M  64
#define SLOTN   (TOPK * T_TOK + NEXP * TILE_M)   // padded slot capacity

// ---------------- optional gfx1250 async global->LDS path --------------------
#if defined(__has_builtin)
#  if __has_builtin(__builtin_amdgcn_global_load_async_to_lds_b128) && \
      __has_builtin(__builtin_amdgcn_s_wait_asynccnt)
#    define MOE_ASYNC_LDS 1
#  endif
#endif
#ifndef MOE_ASYNC_LDS
#  define MOE_ASYNC_LDS 0
#endif

#if MOE_ASYNC_LDS
typedef int moe_v4i __attribute__((ext_vector_type(4)));
typedef moe_v4i __attribute__((address_space(1)))* moe_gp128;   // global b128 ptr
typedef moe_v4i __attribute__((address_space(3)))* moe_lp128;   // LDS    b128 ptr
#  define GLD_ASYNC_B128(gp, lp)                                               \
      __builtin_amdgcn_global_load_async_to_lds_b128(                          \
          (moe_gp128)(gp), (moe_lp128)(lp), 0, 0)
#  define ASYNC_WAIT0() __builtin_amdgcn_s_wait_asynccnt(0)
#else
#  define ASYNC_WAIT0() ((void)0)
#endif

// ---------------- CDNA5 WMMA types ------------------------------------------
typedef __attribute__((ext_vector_type(16))) __bf16 v16bf;
typedef __attribute__((ext_vector_type(8)))  float  v8f;

union Frag16 {
    v16bf v;
    unsigned short s[16];
    unsigned int   u[8];
};

__device__ __forceinline__ unsigned short f2bf(float f) {
    // round-to-nearest-even f32 -> bf16
    unsigned int u = __float_as_uint(f);
    unsigned int r = u + 0x7fffu + ((u >> 16) & 1u);
    return (unsigned short)(r >> 16);
}

// ---------------- 1. init: zero output / slot buffers / accumulators --------
__global__ void moe_init_kernel(float* out, long outN,
                                int* slot_tok, float* slot_w, int slotN,
                                int* counts, float* psum) {
    long i      = (long)blockIdx.x * blockDim.x + threadIdx.x;
    long stride = (long)gridDim.x * blockDim.x;
    for (long k = i; k < outN; k += stride) out[k] = 0.0f;
    for (long k = i; k < slotN; k += stride) { slot_tok[k] = -1; slot_w[k] = 0.0f; }
    if (i < NEXP) { counts[i] = 0; psum[i] = 0.0f; }
}

// ---------------- 2. f32 -> bf16 conversion ----------------------------------
__global__ void conv_bf16_kernel(const float* __restrict__ src,
                                 unsigned short* __restrict__ dst, long n) {
    long i      = (long)blockIdx.x * blockDim.x + threadIdx.x;
    long stride = (long)gridDim.x * blockDim.x;
    for (; i < n; i += stride) dst[i] = f2bf(src[i]);
}

// ---------------- 3. router: logits/softmax/top2 -----------------------------
__global__ __launch_bounds__(256) void router_kernel(
        const float* __restrict__ x, const float* __restrict__ gate_w,
        int* __restrict__ tk_idx, float* __restrict__ tk_w,
        int* counts, float* psum) {
    __shared__ float gws[NEXP * DMODEL];
    __shared__ float psum_l[NEXP];
    for (int i = threadIdx.x; i < NEXP * DMODEL; i += 256) gws[i] = gate_w[i];
    if (threadIdx.x < NEXP) psum_l[threadIdx.x] = 0.0f;
    __syncthreads();

    const int lane = threadIdx.x & 31;
    const int wid  = threadIdx.x >> 5;
    const int gwave  = blockIdx.x * 8 + wid;
    const int nwaves = gridDim.x * 8;

    for (int t = gwave; t < T_TOK; t += nwaves) {
        float xv[8];
#pragma unroll
        for (int i = 0; i < 8; ++i) xv[i] = x[(long)t * DMODEL + lane + 32 * i];

        float pr[NEXP];
#pragma unroll
        for (int e = 0; e < NEXP; ++e) {
            float p = 0.0f;
#pragma unroll
            for (int i = 0; i < 8; ++i) p += xv[i] * gws[e * DMODEL + lane + 32 * i];
#pragma unroll
            for (int m = 16; m >= 1; m >>= 1) p += __shfl_xor(p, m, 32);  // wave32
            pr[e] = p;
        }
        float mx = pr[0];
#pragma unroll
        for (int e = 1; e < NEXP; ++e) mx = fmaxf(mx, pr[e]);
        float se = 0.0f;
#pragma unroll
        for (int e = 0; e < NEXP; ++e) { pr[e] = __expf(pr[e] - mx); se += pr[e]; }
        float inv = 1.0f / se;
#pragma unroll
        for (int e = 0; e < NEXP; ++e) pr[e] *= inv;
        int i1 = 0;
#pragma unroll
        for (int e = 1; e < NEXP; ++e) if (pr[e] > pr[i1]) i1 = e;
        int i2 = (i1 == 0) ? 1 : 0;
#pragma unroll
        for (int e = 0; e < NEXP; ++e) if (e != i1 && pr[e] > pr[i2]) i2 = e;
        float s2 = pr[i1] + pr[i2];

        if (lane == 0) {
            tk_idx[t]       = i1 | (i2 << 8);
            tk_w[2 * t + 0] = pr[i1] / s2;
            tk_w[2 * t + 1] = pr[i2] / s2;
            atomicAdd(&counts[i1], 1);
            atomicAdd(&counts[i2], 1);
#pragma unroll
            for (int e = 0; e < NEXP; ++e) atomicAdd(&psum_l[e], pr[e]);
        }
    }
    __syncthreads();
    if (threadIdx.x < NEXP) atomicAdd(&psum[threadIdx.x], psum_l[threadIdx.x]);
}

// ---------------- 4. offsets: padded exclusive scan --------------------------
__global__ void offsets_kernel(const int* counts, int* off, int* cursors) {
    if (threadIdx.x == 0 && blockIdx.x == 0) {
        int acc = 0;
        for (int e = 0; e < NEXP; ++e) {
            off[e]     = acc;
            cursors[e] = acc;
            acc += (counts[e] + TILE_M - 1) & ~(TILE_M - 1);   // tiles single-expert
        }
        off[NEXP] = acc;
    }
}

// ---------------- 5. scatter tokens into per-expert slot lists ---------------
__global__ void scatter_kernel(const int* __restrict__ tk_idx,
                               const float* __restrict__ tk_w,
                               int* cursors, int* slot_tok, float* slot_w) {
    long i      = (long)blockIdx.x * blockDim.x + threadIdx.x;
    long stride = (long)gridDim.x * blockDim.x;
    for (long t = i; t < T_TOK; t += stride) {
        int pk = tk_idx[t];
#pragma unroll
        for (int k = 0; k < TOPK; ++k) {
            int e   = (pk >> (8 * k)) & 255;
            int pos = atomicAdd(&cursors[e], 1);
            slot_tok[pos] = (int)t;
            slot_w[pos]   = tk_w[2 * t + k];
        }
    }
}

// ---------------- 6. expert GEMMs: WMMA bf16 16x16x32 ------------------------
// block = 256 threads = 8 wave32s; tile = 64 tokens of one expert.
// waves: mw = wid>>1 (M strip of 16 rows), nw = wid&1 (N half).
__global__ __launch_bounds__(256) void expert_kernel(
        const unsigned short* __restrict__ xbf,
        const unsigned short* __restrict__ w1bf,
        const unsigned short* __restrict__ w2bf,
        const float* __restrict__ b1, const float* __restrict__ b2,
        const int* __restrict__ off,
        const int* __restrict__ slot_tok, const float* __restrict__ slot_w,
        float* out) {
    // static LDS ~80.4 KB (CDNA5 WGP has 320 KB)
    __shared__ __align__(16) unsigned short xs[TILE_M * 264];     // X tile  [64][256]+pad
    __shared__ __align__(16) unsigned short wbuf[DMODEL * 72];    // W1 [64][264] | W2 [256][72]
    __shared__ __align__(16) unsigned short hs[TILE_M * 72];      // H chunk [64][64]+pad
    __shared__ int   st[TILE_M];
    __shared__ float swt[TILE_M];
    __shared__ int   s_off[NEXP + 1];

    const int rowStart = blockIdx.x * TILE_M;
    if (threadIdx.x < NEXP + 1) s_off[threadIdx.x] = off[threadIdx.x];
    if (threadIdx.x < TILE_M) {
        st[threadIdx.x]  = slot_tok[rowStart + threadIdx.x];
        swt[threadIdx.x] = slot_w[rowStart + threadIdx.x];
    }
    __syncthreads();

    if (rowStart >= s_off[NEXP]) return;          // uniform across block
    int e = 0;
    while (e < NEXP - 1 && s_off[e + 1] <= rowStart) ++e;

    // ---- gather X tile (bf16), zeros for padded rows ----
#if MOE_ASYNC_LDS
    for (int idx = threadIdx.x; idx < TILE_M * 32; idx += 256) {  // 16B chunks
        int r = idx >> 5, c = idx & 31;
        unsigned short* lp = &xs[r * 264 + c * 8];
        int tok = st[r];
        if (tok >= 0) {
            GLD_ASYNC_B128(xbf + (long)tok * DMODEL + c * 8, lp);
        } else {
            ((unsigned int*)lp)[0] = 0u; ((unsigned int*)lp)[1] = 0u;
            ((unsigned int*)lp)[2] = 0u; ((unsigned int*)lp)[3] = 0u;
        }
    }
#else
    {
        const unsigned int* xb32 = (const unsigned int*)xbf;
        for (int idx = threadIdx.x; idx < TILE_M * 128; idx += 256) {
            int r = idx >> 7, dp = idx & 127;
            int tok = st[r];
            unsigned int v = (tok >= 0) ? xb32[(long)tok * 128 + dp] : 0u;
            *(unsigned int*)&xs[r * 264 + dp * 2] = v;
        }
    }
#endif

    const int lane = threadIdx.x & 31;
    const int wid  = threadIdx.x >> 5;
    const int mw   = wid >> 1;        // 0..3  (M strip of 16 rows)
    const int nw   = wid & 1;         // 0..1  (N half)
    const int half = lane >> 4;
    const int lidx = lane & 15;
    const int r0   = mw * 16;

    v8f yacc[8];
#pragma unroll
    for (int t = 0; t < 8; ++t) { v8f z = {}; yacc[t] = z; }

    for (int h0 = 0; h0 < DHID; h0 += 64) {
        __syncthreads();   // prior fc2 reads of wbuf complete
        // ---- stage W1 chunk: wbuf as [64][264], [n][d] = w1[e][h0+n][d] ----
#if MOE_ASYNC_LDS
        for (int idx = threadIdx.x; idx < 64 * 32; idx += 256) {  // 16B chunks
            int n = idx >> 5, c = idx & 31;
            GLD_ASYNC_B128(w1bf + ((long)e * DHID + h0 + n) * DMODEL + c * 8,
                           &wbuf[n * 264 + c * 8]);
        }
#else
        {
            const unsigned int* w1b32 = (const unsigned int*)w1bf;
            for (int idx = threadIdx.x; idx < 64 * 128; idx += 256) {
                int n = idx >> 7, dp = idx & 127;
                *(unsigned int*)&wbuf[n * 264 + dp * 2] =
                    w1b32[((long)e * DHID + h0 + n) * 128 + dp];
            }
        }
#endif
        if (h0 + 64 < DHID)   // prefetch next W1 chunk (global_prefetch_b8)
            __builtin_prefetch(&w1bf[((long)e * DHID + h0 + 64) * DMODEL], 0, 1);
        ASYNC_WAIT0();
        __syncthreads();

        // ---- fc1: two 16x16 H tiles per wave, K=256, A-frag reused ----
        v8f hacc[2];
        { v8f z = {}; hacc[0] = z; hacc[1] = z; }
#pragma unroll
        for (int k0 = 0; k0 < DMODEL; k0 += 32) {
            Frag16 a;
#pragma unroll
            for (int j = 0; j < 8; ++j) {
                int ka = ((j < 4) ? 0 : 16) + half * 8 + 2 * (j & 3);   // A layout
                a.u[j] = *(const unsigned int*)&xs[(r0 + lidx) * 264 + k0 + ka];
            }
#pragma unroll
            for (int tt = 0; tt < 2; ++tt) {
                int c0 = nw * 32 + tt * 16;
                Frag16 b;
#pragma unroll
                for (int j = 0; j < 8; ++j) {
                    int kb = ((j < 4) ? 0 : 16) + half * 8 + 2 * (j & 3); // B layout
                    b.u[j] = *(const unsigned int*)&wbuf[(c0 + lidx) * 264 + k0 + kb];
                }
                hacc[tt] = __builtin_amdgcn_wmma_f32_16x16x32_bf16(
                               false, a.v, false, b.v, (short)0, hacc[tt], false, false);
            }
        }
        // bias + SiLU -> bf16 H tile in LDS
#pragma unroll
        for (int tt = 0; tt < 2; ++tt) {
            int c0 = nw * 32 + tt * 16;
            float bias = b1[e * DHID + h0 + c0 + lidx];
#pragma unroll
            for (int j = 0; j < 8; ++j) {
                float v  = hacc[tt][j] + bias;
                float sv = v / (1.0f + __expf(-v));
                int   m  = j + half * 8;                                 // C layout
                hs[(r0 + m) * 72 + c0 + lidx] = f2bf(sv);
            }
        }
        __syncthreads();   // hs complete; w1 reads complete -> reuse wbuf

        // ---- stage W2 chunk: wbuf as [256][72], [d][k] = w2[e][d][h0+k] ----
#if MOE_ASYNC_LDS
        for (int idx = threadIdx.x; idx < DMODEL * 8; idx += 256) {   // 16B chunks
            int d = idx >> 3, c = idx & 7;
            GLD_ASYNC_B128(w2bf + ((long)e * DMODEL + d) * DHID + h0 + c * 8,
                           &wbuf[d * 72 + c * 8]);
        }
#else
        {
            const unsigned int* w2b32 = (const unsigned int*)w2bf;
            for (int idx = threadIdx.x; idx < DMODEL * 32; idx += 256) {
                int d = idx >> 5, kp = idx & 31;
                *(unsigned int*)&wbuf[d * 72 + kp * 2] =
                    w2b32[(((long)e * DMODEL + d) * DHID + h0) / 2 + kp];
            }
        }
#endif
        if (h0 + 64 < DHID)
            __builtin_prefetch(&w2bf[(long)e * DMODEL * DHID + h0 + 64], 0, 1);
        ASYNC_WAIT0();
        __syncthreads();

        // ---- fc2: 8 Y tiles per wave over this 64-wide K chunk, A reused ----
#pragma unroll
        for (int k0 = 0; k0 < 64; k0 += 32) {
            Frag16 a;
#pragma unroll
            for (int j = 0; j < 8; ++j) {
                int ka = ((j < 4) ? 0 : 16) + half * 8 + 2 * (j & 3);
                a.u[j] = *(const unsigned int*)&hs[(r0 + lidx) * 72 + k0 + ka];
            }
#pragma unroll
            for (int t = 0; t < 8; ++t) {
                int col0 = nw * 128 + t * 16;
                Frag16 b;
#pragma unroll
                for (int j = 0; j < 8; ++j) {
                    int kb = ((j < 4) ? 0 : 16) + half * 8 + 2 * (j & 3);
                    b.u[j] = *(const unsigned int*)&wbuf[(col0 + lidx) * 72 + k0 + kb];
                }
                yacc[t] = __builtin_amdgcn_wmma_f32_16x16x32_bf16(
                              false, a.v, false, b.v, (short)0, yacc[t], false, false);
            }
        }
    }

    // ---- epilogue: weighted scatter-add (+ b2) into output ----
#pragma unroll
    for (int t = 0; t < 8; ++t) {
        int   col = nw * 128 + t * 16 + lidx;
        float bb2 = b2[e * DMODEL + col];
#pragma unroll
        for (int j = 0; j < 8; ++j) {
            int m   = j + half * 8;
            int row = r0 + m;
            int tok = st[row];
            if (tok >= 0) {
                float w = swt[row];
                atomicAdd(&out[(long)tok * DMODEL + col], w * (yacc[t][j] + bb2));
            }
        }
    }
}

// ---------------- 7. aux loss ------------------------------------------------
__global__ void aux_kernel(const int* counts, const float* psum, float* out_aux) {
    if (threadIdx.x == 0 && blockIdx.x == 0) {
        float s = 0.0f;
        for (int e = 0; e < NEXP; ++e) s += (float)counts[e] * psum[e];
        *out_aux = (float)NEXP * s / ((float)T_TOK * (float)T_TOK);
    }
}

// ---------------- host: launch -----------------------------------------------
extern "C" void kernel_launch(void* const* d_in, const int* in_sizes, int n_in,
                              void* d_out, int out_size, void* d_ws, size_t ws_size,
                              hipStream_t stream) {
    const float* x      = (const float*)d_in[0];   // [T, 256]
    const float* gate_w = (const float*)d_in[1];   // [8, 256]
    const float* w1     = (const float*)d_in[2];   // [8, 512, 256]
    const float* b1     = (const float*)d_in[3];   // [8, 512]
    const float* w2     = (const float*)d_in[4];   // [8, 256, 512]
    const float* b2     = (const float*)d_in[5];   // [8, 256]
    float* out = (float*)d_out;                    // [T*256] out ++ [1] aux

    char*  ws = (char*)d_ws;
    size_t o  = 0;
    auto carve = [&](size_t bytes) -> void* {
        void* p = ws + o;
        o = (o + bytes + 255) & ~(size_t)255;
        return p;
    };
    unsigned short* xbf     = (unsigned short*)carve((size_t)T_TOK * DMODEL * 2);
    unsigned short* w1bf    = (unsigned short*)carve((size_t)NEXP * DHID * DMODEL * 2);
    unsigned short* w2bf    = (unsigned short*)carve((size_t)NEXP * DMODEL * DHID * 2);
    int*            tk_idx  = (int*)carve((size_t)T_TOK * 4);
    float*          tk_w    = (float*)carve((size_t)T_TOK * TOPK * 4);
    int*            slot_tk = (int*)carve((size_t)SLOTN * 4);
    float*          slot_w  = (float*)carve((size_t)SLOTN * 4);
    int*            counts  = (int*)carve(NEXP * 4);
    int*            off     = (int*)carve((NEXP + 1) * 4);
    int*            cursors = (int*)carve(NEXP * 4);
    float*          psum    = (float*)carve(NEXP * 4);

    const long outN = (long)T_TOK * DMODEL + 1;

    moe_init_kernel<<<2048, 256, 0, stream>>>(out, outN, slot_tk, slot_w, SLOTN,
                                              counts, psum);

    conv_bf16_kernel<<<2048, 256, 0, stream>>>(x,  xbf,  (long)T_TOK * DMODEL);
    conv_bf16_kernel<<<512,  256, 0, stream>>>(w1, w1bf, (long)NEXP * DHID * DMODEL);
    conv_bf16_kernel<<<512,  256, 0, stream>>>(w2, w2bf, (long)NEXP * DMODEL * DHID);

    router_kernel<<<1024, 256, 0, stream>>>(x, gate_w, tk_idx, tk_w, counts, psum);
    offsets_kernel<<<1, 32, 0, stream>>>(counts, off, cursors);
    scatter_kernel<<<512, 256, 0, stream>>>(tk_idx, tk_w, cursors, slot_tk, slot_w);

    const int nTiles = SLOTN / TILE_M;   // 4104 tiles; empties exit early
    expert_kernel<<<nTiles, 256, 0, stream>>>(xbf, w1bf, w2bf, b1, b2, off,
                                              slot_tk, slot_w, out);

    aux_kernel<<<1, 32, 0, stream>>>(counts, psum, out + (size_t)T_TOK * DMODEL);
}
